// MSAAttention_6305011990744
// MI455X (gfx1250) — matrix operand order repaired and checked by reading, not compiled, not run on hardware
//
#include <hip/hip_runtime.h>
#include <hip/hip_bf16.h>
#include <math.h>
#include <stdint.h>

// ---------------- problem constants (from reference setup_inputs) -----------
#define S_  64
#define R_  512
#define C_  256
#define H_  8
#define D_  32
#define CZ_ 128
#define SR_ (S_ * R_)          // 32768 rows of the "m" matrix
#define HD_ (H_ * D_)          // 256
#define N4_ (4 * HD_)          // 1024 concat q|k|v|g columns

#define GP_   40               // padded LDS row stride (halves): 80B => bank stride 20
#define VTP_  520              // padded sVt row stride (halves)
#define SCP_  516              // padded sScore row stride (floats): bank stride 4

typedef __attribute__((ext_vector_type(16))) _Float16 v16h;
typedef __attribute__((ext_vector_type(8)))  float    v8f;
typedef __attribute__((ext_vector_type(4)))  unsigned int u32x4;
typedef __attribute__((ext_vector_type(8)))  int      i32x8;
typedef __attribute__((ext_vector_type(4)))  int      i32x4;

// ---- WMMA wave32 fragment layouts (ISA 7.12.2) -----------------------------
// A (16x32 f16): lane l row M=l&15; elements: K runs [8g..8g+7],[16+8g..16+8g+7]
// B (32x16 f16): lane l col N=l&15; elements: K runs [8g..8g+7],[16+8g..16+8g+7]
// C/D (16x16 f32): vgpr i, lane l -> row M=i+8g, col N=l&15

__device__ __forceinline__ v16h load_a_frag(const _Float16* rowp, int g) {
    v16h a;
    ((float4*)&a)[0] = *(const float4*)(rowp + 8 * g);
    ((float4*)&a)[1] = *(const float4*)(rowp + 16 + 8 * g);
    return a;
}
__device__ __forceinline__ v16h load_b_frag(const _Float16* p) {
    const int gb = (threadIdx.x & 31) >> 4;
    v16h b;
    ((float4*)&b)[0] = *(const float4*)(p - 8 * gb);
    ((float4*)&b)[1] = *(const float4*)(p + 16 - 8 * gb);
    return b;
}
__device__ __forceinline__ unsigned lds32(const void* p) {
    // low 32 bits of a generic LDS address are the LDS byte offset (ISA 10.2)
    return (unsigned)(uintptr_t)p;
}

// ---- Tensor Data Mover: 2-D f16 tile (32 elems x tile_h rows) -> LDS -------
// pad=true inserts 16B after every 64B row => LDS row stride 40 halves.
__device__ __forceinline__ void tdm_load_2d_f16(unsigned lds_addr, const void* gp,
                                                int tile_w, int tile_h,
                                                long long row_stride_elems, bool pad) {
    unsigned long long ga = (unsigned long long)(uintptr_t)gp;
    u32x4 g0;
    g0[0] = 1u;                                            // count=1 (valid descriptor)
    g0[1] = lds_addr;
    g0[2] = (unsigned)ga;                                  // global_addr[31:0]
    g0[3] = (unsigned)((ga >> 32) & 0x1FFFFFFu) | (2u << 30); // addr[56:32] | type=2
    i32x8 g1;
    g1[0] = (1 << 16) |                                    // data_size=1 (2 bytes)
            (pad ? ((1 << 20) | (3 << 22) | (3 << 25)) : 0); // pad 16B per 64B
    g1[1] = (tile_w & 0xFFFF) << 16;                       // tensor_dim0[15:0]
    g1[2] = ((unsigned)tile_w >> 16) | ((tile_h & 0xFFFF) << 16);
    g1[3] = ((unsigned)tile_h >> 16) | (tile_w << 16);     // tensor_dim1 hi | tile_dim0
    g1[4] = (tile_h & 0xFFFF);                             // tile_dim1 | tile_dim2=0
    g1[5] = (int)(row_stride_elems & 0xFFFFFFFFll);        // tensor_dim0_stride lo32
    g1[6] = (int)((row_stride_elems >> 32) & 0xFFFFll);
    g1[7] = 0;
    i32x4 z4 = {};
#if defined(__clang_major__) && __clang_major__ >= 23
    i32x8 z8 = {};
    __builtin_amdgcn_tensor_load_to_lds(g0, g1, z4, z4, z8, 0);
#else
    __builtin_amdgcn_tensor_load_to_lds(g0, g1, z4, z4, 0);
#endif
}

// ---------------------------------------------------------------------------
// K0: pack weights f16 TRANSPOSED (N-major) so GEMM B tiles are K-contiguous.
// ---------------------------------------------------------------------------
__global__ void k_pack_weights(const float* __restrict__ wq, const float* __restrict__ wk,
                               const float* __restrict__ wv, const float* __restrict__ wg,
                               const float* __restrict__ wo,
                               _Float16* __restrict__ WcatT, _Float16* __restrict__ WoT) {
    const int total = N4_ * C_ + C_ * HD_;
    for (int idx = blockIdx.x * blockDim.x + threadIdx.x; idx < total;
         idx += gridDim.x * blockDim.x) {
        if (idx < N4_ * C_) {
            int col = idx >> 8, c = idx & (C_ - 1);        // WcatT[col][c]
            int blk = col >> 8, hd = col & (HD_ - 1);
            const float* w = (blk == 0) ? wq : (blk == 1) ? wk : (blk == 2) ? wv : wg;
            float v = w[c * HD_ + hd];
            if (blk == 0) v *= 0.17677669529663689f;       // fold 1/sqrt(32) into Q
            WcatT[idx] = (_Float16)v;
        } else {
            int j = idx - N4_ * C_;
            int c = j >> 8, hd = j & (HD_ - 1);            // WoT[c][hd] = wo[hd][c]
            WoT[j] = (_Float16)wo[hd * C_ + c];
        }
    }
}

// ---------------------------------------------------------------------------
// K1: zb[h][q][k] = sum_z LN(z[q,k,:]) * w_z[:,h].  One wave per (q,k) row.
// ---------------------------------------------------------------------------
__global__ void k_ln_z_bias(const float* __restrict__ z, const float* __restrict__ gw,
                            const float* __restrict__ gb, const float* __restrict__ w_z,
                            float* __restrict__ zb) {
    const int wv = threadIdx.x >> 5, lane = threadIdx.x & 31;
    const int row = blockIdx.x * 8 + wv;                   // [0, R*R)
    const float* zr = z + (size_t)row * CZ_;
    const int c = lane * 4;
    float4 x = *reinterpret_cast<const float4*>(zr + c);
    float s  = x.x + x.y + x.z + x.w;
    float ss = x.x * x.x + x.y * x.y + x.z * x.z + x.w * x.w;
    for (int o = 16; o; o >>= 1) { s += __shfl_xor(s, o); ss += __shfl_xor(ss, o); }
    const float mu = s * (1.0f / CZ_);
    const float rs = rsqrtf(ss * (1.0f / CZ_) - mu * mu + 1e-5f);
    float l0 = (x.x - mu) * rs * gw[c + 0] + gb[c + 0];
    float l1 = (x.y - mu) * rs * gw[c + 1] + gb[c + 1];
    float l2 = (x.z - mu) * rs * gw[c + 2] + gb[c + 2];
    float l3 = (x.w - mu) * rs * gw[c + 3] + gb[c + 3];
    float acc[H_];
#pragma unroll
    for (int h = 0; h < H_; ++h)
        acc[h] = l0 * w_z[(c + 0) * H_ + h] + l1 * w_z[(c + 1) * H_ + h] +
                 l2 * w_z[(c + 2) * H_ + h] + l3 * w_z[(c + 3) * H_ + h];
#pragma unroll
    for (int h = 0; h < H_; ++h) {
        float v = acc[h];
        for (int o = 16; o; o >>= 1) v += __shfl_xor(v, o);
        if (lane == 0) zb[h * (R_ * R_) + row] = v;
    }
}

// ---------------------------------------------------------------------------
// K2: mn = LN(m) as f16 [SR][C].  One wave per row.
// ---------------------------------------------------------------------------
__global__ void k_ln_m(const float* __restrict__ m, const float* __restrict__ gw,
                       const float* __restrict__ gb, _Float16* __restrict__ mn) {
    const int wv = threadIdx.x >> 5, lane = threadIdx.x & 31;
    const int row = blockIdx.x * 8 + wv;                   // [0, SR)
    const float* mr = m + (size_t)row * C_;
    const int c = lane * 8;
    float4 x0 = *reinterpret_cast<const float4*>(mr + c);
    float4 x1 = *reinterpret_cast<const float4*>(mr + c + 4);
    float s  = x0.x + x0.y + x0.z + x0.w + x1.x + x1.y + x1.z + x1.w;
    float ss = x0.x * x0.x + x0.y * x0.y + x0.z * x0.z + x0.w * x0.w +
               x1.x * x1.x + x1.y * x1.y + x1.z * x1.z + x1.w * x1.w;
    for (int o = 16; o; o >>= 1) { s += __shfl_xor(s, o); ss += __shfl_xor(ss, o); }
    const float mu = s * (1.0f / C_);
    const float rs = rsqrtf(ss * (1.0f / C_) - mu * mu + 1e-5f);
    float xv[8] = {x0.x, x0.y, x0.z, x0.w, x1.x, x1.y, x1.z, x1.w};
#pragma unroll
    for (int j = 0; j < 8; ++j)
        mn[(size_t)row * C_ + c + j] = (_Float16)((xv[j] - mu) * rs * gw[c + j] + gb[c + j]);
}

// ---------------------------------------------------------------------------
// K3: f16 WMMA GEMM, B TRANSPOSED: C[M,N] = A[M,K]*Bt[N,K]^T (+bias).
//     TDM-staged double-buffered tiles; B-fragments software-pipelined.
// ---------------------------------------------------------------------------
__global__ void k_gemm_wmma(const _Float16* __restrict__ A, const _Float16* __restrict__ Bt,
                            void* __restrict__ Cout, int M, int N, int K,
                            int out_f16, const float* __restrict__ bias) {
    __shared__ _Float16 sA [2][128 * GP_];                 // [m][k], 80B rows
    __shared__ _Float16 sBt[2][128 * GP_];                 // [n][k]
    const int t = threadIdx.x, wv = t >> 5, lane = t & 31;
    const int g = lane >> 4, ln = lane & 15;
    const int m0 = blockIdx.y * 128, n0 = blockIdx.x * 128;

    v8f acc[8];
#pragma unroll
    for (int i = 0; i < 8; ++i) acc[i] = (v8f){};

    if (wv == 0) {                                         // prologue DMA (tile 0)
        tdm_load_2d_f16(lds32(&sA [0][0]), &A [(size_t)m0 * K], 32, 128, K, true);
        tdm_load_2d_f16(lds32(&sBt[0][0]), &Bt[(size_t)n0 * K], 32, 128, K, true);
        __builtin_amdgcn_s_wait_tensorcnt(0);
    }
    __syncthreads();

    int p = 0;
    for (int kt = 0; kt < K; kt += 32) {
        const bool more = (kt + 32 < K);
        if (more && wv == 0) {                             // DMA next tile, other buffer
            tdm_load_2d_f16(lds32(&sA [p ^ 1][0]), &A [(size_t)m0 * K + kt + 32], 32, 128, K, true);
            tdm_load_2d_f16(lds32(&sBt[p ^ 1][0]), &Bt[(size_t)n0 * K + kt + 32], 32, 128, K, true);
        }
        v16h af = load_a_frag(&sA[p][(wv * 16 + ln) * GP_], g);
        v16h bf = load_b_frag(&sBt[p][ln * GP_ + g * 16]); // nt = 0
#pragma unroll
        for (int nt = 0; nt < 8; ++nt) {
            v16h bfn;
            if (nt < 7)                                    // pipeline next B fragment
                bfn = load_b_frag(&sBt[p][((nt + 1) * 16 + ln) * GP_ + g * 16]);
            acc[nt] = __builtin_amdgcn_wmma_f32_16x16x32_f16(
                false, af, false, bf, (short)0, acc[nt], false, false);
            bf = bfn;
        }
        if (more) {
            if (wv == 0) __builtin_amdgcn_s_wait_tensorcnt(0);
            __syncthreads();
            p ^= 1;
        }
    }

    float bvals[8] = {0.f, 0.f, 0.f, 0.f, 0.f, 0.f, 0.f, 0.f};
    if (bias) {
#pragma unroll
        for (int nt = 0; nt < 8; ++nt) bvals[nt] = bias[n0 + nt * 16 + ln];
    }
#pragma unroll
    for (int nt = 0; nt < 8; ++nt) {
#pragma unroll
        for (int i = 0; i < 8; ++i) {
            int row = m0 + wv * 16 + i + 8 * g;
            int col = n0 + nt * 16 + ln;
            float v = acc[nt][i] + bvals[nt];
            if (out_f16) ((_Float16*)Cout)[(size_t)row * N + col] = (_Float16)v;
            else         ((float*)Cout)[(size_t)row * N + col] = v;
        }
    }
}

// ---------------------------------------------------------------------------
// K4: attention for one (s, h, 64-query tile).  K & Q tiles via TDM (padded
//     rows); score WMMA loop is pure LDS; zb + mask bias folded into softmax
//     pass 1 as coalesced float4 reads.
// ---------------------------------------------------------------------------
__global__ void k_attention(const _Float16* __restrict__ qkvg, const float* __restrict__ zb,
                            const float* __restrict__ mask, float* __restrict__ O) {
    extern __shared__ char smem[];
    float*    sScore = (float*)smem;                       // [64][SCP_]
    float*    sMB    = sScore + 64 * SCP_;                 // [512]
    _Float16* sK     = (_Float16*)(sMB + R_);              // [512][GP_] key-major
    _Float16* sVt    = sK + R_ * GP_;                      // [32][VTP_] d-major
    _Float16* sQ     = sVt + D_ * VTP_;                    // [64][GP_]

    const int t = threadIdx.x, wv = t >> 5, lane = t & 31;
    const int g = lane >> 4, ln = lane & 15;
    const int qt = blockIdx.x, h = blockIdx.y, s = blockIdx.z;
    const int q0 = qt * 64;
    const size_t rowbase = (size_t)s * R_;

    // ---- TDM: DMA K [512x32] and Q [64x32] tiles into padded LDS ----------
    if (wv == 0) {
        tdm_load_2d_f16(lds32(sK), qkvg + rowbase * N4_ + HD_ + h * D_, D_, R_, N4_, true);
        tdm_load_2d_f16(lds32(sQ), qkvg + (rowbase + q0) * N4_ + h * D_, D_, 64, N4_, true);
    }
    // ---- V: load rows, store transposed; mask bias (overlaps the DMA) -----
    for (int r = t; r < R_; r += 256) {
        float4 buf[4];
        const float4* vsrc = (const float4*)&qkvg[(rowbase + r) * N4_ + 2 * HD_ + h * D_];
        buf[0] = vsrc[0]; buf[1] = vsrc[1]; buf[2] = vsrc[2]; buf[3] = vsrc[3];
        const _Float16* hb = (const _Float16*)buf;
#pragma unroll
        for (int d = 0; d < D_; ++d) sVt[d * VTP_ + r] = hb[d];
        sMB[r] = 1e9f * (mask[rowbase + r] - 1.0f);
    }
    if (wv == 0) __builtin_amdgcn_s_wait_tensorcnt(0);
    __syncthreads();

    // ---- scores: wave w -> 16 q-rows (w&3), 256 keys (w>>2); pure LDS -----
    {
        const int qr = (wv & 3) * 16;
        const int kb0 = (wv >> 2) * 256;
        v16h af = load_a_frag(&sQ[(qr + ln) * GP_], g);
        v16h bf = load_b_frag(&sK[(kb0 + ln) * GP_ + g * 16]);
        for (int kt = 0; kt < 16; ++kt) {
            const int kb = kb0 + kt * 16;
            v16h bfn;
            if (kt < 15)                                   // pipeline next K fragment
                bfn = load_b_frag(&sK[(kb + 16 + ln) * GP_ + g * 16]);
            v8f acc = (v8f){};
            acc = __builtin_amdgcn_wmma_f32_16x16x32_f16(
                false, af, false, bf, (short)0, acc, false, false);
            const int key = kb + ln;
#pragma unroll
            for (int i = 0; i < 8; ++i)
                sScore[(qr + i + 8 * g) * SCP_ + key] = acc[i];
            bf = bfn;
        }
    }
    __syncthreads();

    // ---- softmax (+zb +mask bias): 4 threads/row, float4 traffic ----------
    {
        const int row = t >> 2, qd = t & 3;
        float4* rp = (float4*)(sScore + row * SCP_) + qd * 32;
        const float4* zp = (const float4*)(zb + h * (R_ * R_) + (size_t)(q0 + row) * R_) + qd * 32;
        const float4* mb = (const float4*)sMB + qd * 32;
        float mx = -1e30f;
        for (int k = 0; k < 32; ++k) {                     // pass 1: bias + max
            float4 v = rp[k], zv = zp[k], mv = mb[k];
            v.x += zv.x + mv.x; v.y += zv.y + mv.y;
            v.z += zv.z + mv.z; v.w += zv.w + mv.w;
            rp[k] = v;
            mx = fmaxf(mx, fmaxf(fmaxf(v.x, v.y), fmaxf(v.z, v.w)));
        }
        mx = fmaxf(mx, __shfl_xor(mx, 1));
        mx = fmaxf(mx, __shfl_xor(mx, 2));
        float sum = 0.0f;
        for (int k = 0; k < 32; ++k) {                     // pass 2: exp + sum
            float4 v = rp[k];
            v.x = __expf(v.x - mx); v.y = __expf(v.y - mx);
            v.z = __expf(v.z - mx); v.w = __expf(v.w - mx);
            sum += v.x + v.y + v.z + v.w;
            rp[k] = v;
        }
        sum += __shfl_xor(sum, 1);
        sum += __shfl_xor(sum, 2);
        const float inv = 1.0f / sum;
        for (int k = 0; k < 32; ++k) {                     // pass 3: normalize
            float4 v = rp[k];
            v.x *= inv; v.y *= inv; v.z *= inv; v.w *= inv;
            rp[k] = v;
        }
    }
    __syncthreads();

    // ---- O = P @ V : waves 0..3, 32-key steps, stage-pipelined ------------
    if (wv < 4) {
        const int qr = wv * 16;
        const float* srow0 = &sScore[(qr + ln) * SCP_];
        v8f acc0 = (v8f){}, acc1 = (v8f){};
        float4 p0, p1, p2, p3;
        v16h b0, b1;
        {   // prologue loads (kt = 0)
            const float* srow = srow0;
            p0 = *(const float4*)(srow + 8 * g);
            p1 = *(const float4*)(srow + 8 * g + 4);
            p2 = *(const float4*)(srow + 16 + 8 * g);
            p3 = *(const float4*)(srow + 16 + 8 * g + 4);
            b0 = load_b_frag(&sVt[ln * VTP_ + g * 16]);
            b1 = load_b_frag(&sVt[(16 + ln) * VTP_ + g * 16]);
        }
        for (int kt = 0; kt < 16; ++kt) {
            float4 q0v, q1v, q2v, q3v;
            v16h c0, c1;
            if (kt < 15) {                                 // pipeline next stage
                const int k0n = (kt + 1) * 32;
                const float* srow = srow0 + k0n;
                q0v = *(const float4*)(srow + 8 * g);
                q1v = *(const float4*)(srow + 8 * g + 4);
                q2v = *(const float4*)(srow + 16 + 8 * g);
                q3v = *(const float4*)(srow + 16 + 8 * g + 4);
                c0 = load_b_frag(&sVt[ln * VTP_ + k0n + g * 16]);
                c1 = load_b_frag(&sVt[(16 + ln) * VTP_ + k0n + g * 16]);
            }
            v16h af;
            af[0] = (_Float16)p0.x; af[1] = (_Float16)p0.y;
            af[2] = (_Float16)p0.z; af[3] = (_Float16)p0.w;
            af[4] = (_Float16)p1.x; af[5] = (_Float16)p1.y;
            af[6] = (_Float16)p1.z; af[7] = (_Float16)p1.w;
            af[8] = (_Float16)p2.x; af[9] = (_Float16)p2.y;
            af[10] = (_Float16)p2.z; af[11] = (_Float16)p2.w;
            af[12] = (_Float16)p3.x; af[13] = (_Float16)p3.y;
            af[14] = (_Float16)p3.z; af[15] = (_Float16)p3.w;
            acc0 = __builtin_amdgcn_wmma_f32_16x16x32_f16(
                false, af, false, b0, (short)0, acc0, false, false);
            acc1 = __builtin_amdgcn_wmma_f32_16x16x32_f16(
                false, af, false, b1, (short)0, acc1, false, false);
            p0 = q0v; p1 = q1v; p2 = q2v; p3 = q3v; b0 = c0; b1 = c1;
        }
#pragma unroll
        for (int i = 0; i < 8; ++i) {
            int ql = qr + i + 8 * g;
            size_t orow = (rowbase + q0 + ql) * HD_;
            O[orow + h * D_ + ln]      = acc0[i];
            O[orow + h * D_ + 16 + ln] = acc1[i];
        }
    }
}

// ---------------------------------------------------------------------------
// K5: gated output X = O * sigmoid(gpre + bg)  (f16), 4 elements per thread
// ---------------------------------------------------------------------------
__global__ void k_gate(const _Float16* __restrict__ qkvg, const float* __restrict__ bg,
                       const float* __restrict__ O, _Float16* __restrict__ X) {
    int idx4 = blockIdx.x * blockDim.x + threadIdx.x;      // [0, SR*HD/4)
    int row = idx4 >> 6, hd4 = (idx4 & 63) * 4;
    float4 o = ((const float4*)O)[idx4];
    float4 b = *(const float4*)&bg[hd4];
    const _Float16* gp = &qkvg[(size_t)row * N4_ + 3 * HD_ + hd4];
    float g0 = (float)gp[0] + b.x, g1 = (float)gp[1] + b.y;
    float g2 = (float)gp[2] + b.z, g3 = (float)gp[3] + b.w;
    _Float16 outv[4];
    outv[0] = (_Float16)(o.x / (1.0f + __expf(-g0)));
    outv[1] = (_Float16)(o.y / (1.0f + __expf(-g1)));
    outv[2] = (_Float16)(o.z / (1.0f + __expf(-g2)));
    outv[3] = (_Float16)(o.w / (1.0f + __expf(-g3)));
    *(float2*)&X[(size_t)idx4 * 4] = *(const float2*)outv;
}

// ---------------------------------------------------------------------------
extern "C" void kernel_launch(void* const* d_in, const int* in_sizes, int n_in,
                              void* d_out, int out_size, void* d_ws, size_t ws_size,
                              hipStream_t stream) {
    const float* m      = (const float*)d_in[0];
    const float* z      = (const float*)d_in[1];
    const float* mask   = (const float*)d_in[2];
    const float* ln_m_w = (const float*)d_in[3];
    const float* ln_m_b = (const float*)d_in[4];
    const float* ln_z_w = (const float*)d_in[5];
    const float* ln_z_b = (const float*)d_in[6];
    const float* w_z    = (const float*)d_in[7];
    const float* wq     = (const float*)d_in[8];
    const float* wk     = (const float*)d_in[9];
    const float* wv     = (const float*)d_in[10];
    const float* wg     = (const float*)d_in[11];
    const float* bg     = (const float*)d_in[12];
    const float* wo     = (const float*)d_in[13];
    const float* bo     = (const float*)d_in[14];

    char* ws = (char*)d_ws;
    _Float16* WcatT = (_Float16*)(ws);                                 // 512 KB
    _Float16* WoT   = (_Float16*)(ws + (512u << 10));                  // 128 KB
    _Float16* mn    = (_Float16*)(ws + (640u << 10));                  // 16 MB (reused as X)
    _Float16* qkvg  = (_Float16*)(ws + (640u << 10) + (size_t)SR_ * C_ * 2);   // 64 MB
    float*    zbp   = (float*)((char*)qkvg + (size_t)SR_ * N4_ * 2);   // 8 MB
    float*    Optr  = (float*)((char*)zbp + (size_t)H_ * R_ * R_ * 4); // 32 MB
    _Float16* X     = mn;

    k_pack_weights<<<512, 256, 0, stream>>>(wq, wk, wv, wg, wo, WcatT, WoT);
    k_ln_z_bias<<<(R_ * R_) / 8, 256, 0, stream>>>(z, ln_z_w, ln_z_b, w_z, zbp);
    k_ln_m<<<SR_ / 8, 256, 0, stream>>>(m, ln_m_w, ln_m_b, mn);
    k_gemm_wmma<<<dim3(N4_ / 128, SR_ / 128), 256, 0, stream>>>(
        mn, WcatT, qkvg, SR_, N4_, C_, /*out_f16=*/1, nullptr);
    const size_t attn_lds = 64 * SCP_ * 4 + R_ * 4 + R_ * GP_ * 2 + D_ * VTP_ * 2 + 64 * GP_ * 2;
    k_attention<<<dim3(R_ / 64, H_, S_), 256, attn_lds, stream>>>(qkvg, zbp, mask, Optr);
    k_gate<<<(SR_ * HD_) / (256 * 4), 256, 0, stream>>>(qkvg, bg, Optr, X);
    k_gemm_wmma<<<dim3(HD_ / 128, SR_ / 128), 256, 0, stream>>>(
        X, WoT, d_out, SR_, C_, HD_, /*out_f16=*/0, bo);
}
